// VQVAE_81157702025646
// MI455X (gfx1250) — matrix operand order repaired and checked by reading, not compiled, not run on hardware
//
#include <hip/hip_runtime.h>
#include <hip/hip_bf16.h>

// VQ-VAE forward for MI455X (gfx1250, wave32, WMMA).
// Fused encoder keeps h1/h2 tiles in LDS (avoids ~0.5-1 GB of f32 intermediate
// HBM traffic; mandatory traffic is only ~25 MB at 23.3 TB/s). conv2/conv3 and
// the final 64x24576 decoder GEMM use v_wmma_f32_16x16x32_f16.

typedef __attribute__((ext_vector_type(16))) _Float16 v16h;
typedef __attribute__((ext_vector_type(8)))  _Float16 v8h;
typedef __attribute__((ext_vector_type(8)))  float    v8f;

#define WMMA_F16(a, b, c) \
    __builtin_amdgcn_wmma_f32_16x16x32_f16(false, (a), false, (b), (short)0, (c), false, false)

__device__ __forceinline__ float lrelu(float v) { return v > 0.f ? v : 0.001f * v; }

// Monotone float <-> uint mapping so we can use integer atomicMax for the pool.
__device__ __forceinline__ unsigned encodeF(float x) {
    unsigned u = __float_as_uint(x);
    return (u & 0x80000000u) ? ~u : (u | 0x80000000u);
}
__device__ __forceinline__ float decodeF(unsigned u) {
    u = (u & 0x80000000u) ? (u & 0x7FFFFFFFu) : ~u;
    return __uint_as_float(u);
}

// A fragment (16x32 f16): lane m = l&15; K halves = (l>>4)*16 + {0..15}.
__device__ __forceinline__ v16h ldsA(const _Float16* A, int ldA, int row0, int k0, int lane) {
    const int m  = lane & 15;
    const int kb = (lane >> 4) << 4;
    const _Float16* p = A + (row0 + m) * ldA + k0 + kb;
    v8h lo = *(const v8h*)p;
    v8h hi = *(const v8h*)(p + 8);
    return __builtin_shufflevector(lo, hi, 0, 1, 2, 3, 4, 5, 6, 7, 8, 9, 10, 11, 12, 13, 14, 15);
}

// Same A layout but sourced from global f32 (row-major, stride ldA).
__device__ __forceinline__ v16h glbA(const float* A, int ldA, int row0, int k0, int lane) {
    const int m  = lane & 15;
    const int kb = (lane >> 4) << 4;
    const float* p = A + (size_t)(row0 + m) * ldA + k0 + kb;
    v16h a;
#pragma unroll
    for (int i = 0; i < 8; ++i) {
        a[i]     = (_Float16)p[i];
        a[8 + i] = (_Float16)p[8 + i];
    }
    return a;
}

// B fragment (32x16 f16) from LDS stored N-major ([n][k], stride ldB):
// lane n = l&15 holds 16 consecutive K halves at k0 + (l>>4)*16.
__device__ __forceinline__ v16h ldsBt(const _Float16* Bt, int ldB, int col0, int k0, int lane) {
    const int n  = lane & 15;
    const int kb = (lane >> 4) << 4;
    const _Float16* p = Bt + (col0 + n) * ldB + k0 + kb;
    v8h lo = *(const v8h*)p;
    v8h hi = *(const v8h*)(p + 8);
    return __builtin_shufflevector(lo, hi, 0, 1, 2, 3, 4, 5, 6, 7, 8, 9, 10, 11, 12, 13, 14, 15);
}

// ---------------------------------------------------------------------------
// Kernel 0: zero the encoded-max buffer and the loss accumulator.
// ---------------------------------------------------------------------------
__global__ void vq_init(unsigned* __restrict__ zbits, float* __restrict__ lossp) {
    const int i = blockIdx.x * 256 + threadIdx.x;
    if (i < 64 * 128) zbits[i] = 0u;   // encode(-inf) > 0, so 0 never wins the max
    if (i == 0) *lossp = 0.f;
}

// ---------------------------------------------------------------------------
// Kernel 1: fused encoder. One block = (batch b, 128-point tile).
//   conv1 (3->64) in VALU, conv2 (64->128) + conv3 (128->128) via WMMA,
//   fused max-pool over points with atomicMax into zbits.
// ---------------------------------------------------------------------------
__global__ __launch_bounds__(256) void vq_encoder(
    const float* __restrict__ x,
    const float* __restrict__ w1, const float* __restrict__ b1,
    const float* __restrict__ w2, const float* __restrict__ b2,
    const float* __restrict__ w3, const float* __restrict__ b3,
    unsigned* __restrict__ zbits)
{
    __shared__ __align__(16) float    sX[3 * 128];
    __shared__ float                  sW1[64 * 3];
    __shared__ float                  sB1[64];
    __shared__ float                  sB2[128];
    __shared__ float                  sB3[128];
    __shared__ __align__(16) _Float16 sH1t[128 * 64];    // [t][c]  (t-major => B frags contiguous)
    __shared__ __align__(16) _Float16 sH2t[128 * 128];   // [t][o]

    const int tid = threadIdx.x;
    const int b   = blockIdx.y;
    const int n0  = blockIdx.x * 128;

    for (int i = tid; i < 3 * 128; i += 256) {
        const int ch = i >> 7, t = i & 127;
        sX[i] = x[((size_t)b * 3 + ch) * 8192 + n0 + t];
    }
    if (tid < 192) sW1[tid] = w1[tid];
    if (tid < 64)  sB1[tid] = b1[tid];
    if (tid < 128) { sB2[tid] = b2[tid]; sB3[tid] = b3[tid]; }
    __syncthreads();

    // conv1: 3 -> 64, LeakyReLU, stored [t][c] as f16.
    for (int i = tid; i < 128 * 64; i += 256) {
        const int t = i >> 6, c = i & 63;
        float acc = sB1[c]
                  + sW1[c * 3 + 0] * sX[t]
                  + sW1[c * 3 + 1] * sX[128 + t]
                  + sW1[c * 3 + 2] * sX[256 + t];
        sH1t[i] = (_Float16)lrelu(acc);
    }
    __syncthreads();

    const int wave = tid >> 5;
    const int lane = tid & 31;
    const int orow = wave * 16;          // this wave's 16 output channels
    const int lg   = (lane >> 4) << 3;   // C-layout row offset (0 or 8)

    // ---- conv2: h2[o,t] = lrelu(W2 @ h1 + b2), K = 64 (2 chained WMMAs) ----
    {
        const v16h a0 = glbA(w2, 64, orow, 0,  lane);
        const v16h a1 = glbA(w2, 64, orow, 32, lane);
        float bias2[8];
#pragma unroll
        for (int r = 0; r < 8; ++r) bias2[r] = sB2[orow + lg + r];
#pragma unroll
        for (int tt = 0; tt < 8; ++tt) {
            v16h bf0 = ldsBt(sH1t, 64, tt * 16, 0,  lane);
            v16h bf1 = ldsBt(sH1t, 64, tt * 16, 32, lane);
            v8f  c   = {};
            c = WMMA_F16(a0, bf0, c);
            c = WMMA_F16(a1, bf1, c);
            const int tcol = tt * 16 + (lane & 15);
            v8h hv;
#pragma unroll
            for (int r = 0; r < 8; ++r) hv[r] = (_Float16)lrelu(c[r] + bias2[r]);
            *(v8h*)(&sH2t[tcol * 128 + orow + lg]) = hv;
        }
    }
    __syncthreads();

    // ---- conv3: h3 = W3 @ h2 + b3 (no activation), K = 128, fused max over t ----
    {
        v16h a3[4];
#pragma unroll
        for (int kk = 0; kk < 4; ++kk) a3[kk] = glbA(w3, 128, orow, kk * 32, lane);
        float bias3[8], mx[8];
#pragma unroll
        for (int r = 0; r < 8; ++r) { bias3[r] = sB3[orow + lg + r]; mx[r] = -INFINITY; }
#pragma unroll
        for (int tt = 0; tt < 8; ++tt) {
            v8f c = {};
#pragma unroll
            for (int kk = 0; kk < 4; ++kk)
                c = WMMA_F16(a3[kk], ldsBt(sH2t, 128, tt * 16, kk * 32, lane), c);
#pragma unroll
            for (int r = 0; r < 8; ++r) mx[r] = fmaxf(mx[r], c[r] + bias3[r]);
        }
        // Reduce over the 16 point-columns (lanes within each 16-lane half).
#pragma unroll
        for (int off = 1; off < 16; off <<= 1) {
#pragma unroll
            for (int r = 0; r < 8; ++r) mx[r] = fmaxf(mx[r], __shfl_xor(mx[r], off, 32));
        }
        if ((lane & 15) == 0) {
#pragma unroll
            for (int r = 0; r < 8; ++r)
                atomicMax(&zbits[b * 128 + orow + lg + r], encodeF(mx[r]));
        }
    }
}

// ---------------------------------------------------------------------------
// Kernel 2: VQ argmin over K=8192 codes + loss + decoder layers 1&2.
// One block per batch row. ~0.13 GF total: VALU with float4 loads.
// ---------------------------------------------------------------------------
__global__ __launch_bounds__(256) void vq_assign_dec12(
    const unsigned* __restrict__ zbits,
    const float* __restrict__ codebook,
    const float* __restrict__ dw1, const float* __restrict__ db1,
    const float* __restrict__ dw2, const float* __restrict__ db2,
    float* __restrict__ zout, float* __restrict__ lossp,
    float* __restrict__ g2buf)
{
    __shared__ __align__(16) float sZ[128];
    __shared__ float               sDist[256];
    __shared__ int                 sIdx[256];
    __shared__ __align__(16) float sZq[128];
    __shared__ float               sG1[128];

    const int tid = threadIdx.x;
    const int b   = blockIdx.x;

    if (tid < 128) {
        const float zv = decodeF(zbits[b * 128 + tid]);
        sZ[tid] = zv;
        zout[b * 128 + tid] = zv;        // z output section
    }
    __syncthreads();

    // d(k) = ||e_k||^2 - 2 z.e_k  (||z||^2 is constant per row, irrelevant to argmin)
    const float4* z4 = (const float4*)sZ;
    float best = INFINITY; int bidx = 0x7FFFFFFF;
    for (int k = tid; k < 8192; k += 256) {
        const float4* cr = (const float4*)(codebook + (size_t)k * 128);
        float s = 0.f;
#pragma unroll 8
        for (int j = 0; j < 32; ++j) {
            const float4 c4 = cr[j], zz = z4[j];
            s += c4.x * (c4.x - 2.f * zz.x);
            s += c4.y * (c4.y - 2.f * zz.y);
            s += c4.z * (c4.z - 2.f * zz.z);
            s += c4.w * (c4.w - 2.f * zz.w);
        }
        if (s < best || (s == best && k < bidx)) { best = s; bidx = k; }
    }
    sDist[tid] = best; sIdx[tid] = bidx;
    __syncthreads();
    for (int stride = 128; stride > 0; stride >>= 1) {
        if (tid < stride) {
            const float od = sDist[tid + stride]; const int oi = sIdx[tid + stride];
            if (od < sDist[tid] || (od == sDist[tid] && oi < sIdx[tid])) {
                sDist[tid] = od; sIdx[tid] = oi;
            }
        }
        __syncthreads();
    }
    const int idx = sIdx[0];

    // z_q gather + loss contribution: 1.25 * sum((z_q - z)^2) / (B*E)
    if (tid < 128) {
        const float q = codebook[(size_t)idx * 128 + tid];
        sZq[tid] = q;
        const float d = q - sZ[tid];
        sDist[tid] = d * d;
    } else {
        sDist[tid] = 0.f;
    }
    __syncthreads();
    for (int stride = 128; stride > 0; stride >>= 1) {
        if (tid < stride) sDist[tid] += sDist[tid + stride];
        __syncthreads();
    }
    if (tid == 0) atomicAdd(lossp, sDist[0] * (1.25f / 8192.f));

    // decoder layer 1: 128 -> 128, LeakyReLU  (z_st == z_q in the forward pass)
    if (tid < 128) {
        const float* wr = dw1 + (size_t)tid * 128;
        float acc = db1[tid];
        for (int e = 0; e < 128; ++e) acc += wr[e] * sZq[e];
        sG1[tid] = lrelu(acc);
    }
    __syncthreads();
    // decoder layer 2: 128 -> 64, LeakyReLU
    if (tid < 64) {
        const float* wr = dw2 + (size_t)tid * 128;
        float acc = db2[tid];
        for (int e = 0; e < 128; ++e) acc += wr[e] * sG1[e];
        g2buf[b * 64 + tid] = lrelu(acc);
    }
}

// ---------------------------------------------------------------------------
// Kernel 3: x_recon = g2[64,64] @ dec_w3.T[64,24576] + b3 via WMMA (K=64).
// dec_w3 is [24576,64] row-major == exactly the [n][k] B-fragment layout.
// Each block covers 32 output columns; 8 waves = 4 M-tiles x 2 N-tiles.
// ---------------------------------------------------------------------------
__global__ __launch_bounds__(256) void vq_decoder3(
    const float* __restrict__ g2buf,
    const float* __restrict__ dw3, const float* __restrict__ db3,
    float* __restrict__ out)
{
    __shared__ __align__(16) _Float16 sG2[64 * 64];
    const int tid = threadIdx.x;
    for (int i = tid; i < 64 * 64; i += 256) sG2[i] = (_Float16)g2buf[i];
    __syncthreads();

    const int wave  = tid >> 5, lane = tid & 31;
    const int mtile = wave & 3, ntile = wave >> 2;
    const int ncol  = blockIdx.x * 32 + ntile * 16 + (lane & 15);

    const v16h a0 = ldsA(sG2, 64, mtile * 16, 0,  lane);
    const v16h a1 = ldsA(sG2, 64, mtile * 16, 32, lane);

    const float* p = dw3 + (size_t)ncol * 64 + ((lane >> 4) << 4);
    v16h bf0, bf1;
#pragma unroll
    for (int i = 0; i < 16; ++i) {
        bf0[i] = (_Float16)p[i];
        bf1[i] = (_Float16)p[32 + i];
    }

    v8f c = {};
    c = WMMA_F16(a0, bf0, c);
    c = WMMA_F16(a1, bf1, c);

    const float bias  = db3[ncol];
    const int   mbase = mtile * 16 + ((lane >> 4) << 3);
#pragma unroll
    for (int r = 0; r < 8; ++r)
        out[(size_t)(mbase + r) * 24576 + ncol] = c[r] + bias;
}

// ---------------------------------------------------------------------------
extern "C" void kernel_launch(void* const* d_in, const int* in_sizes, int n_in,
                              void* d_out, int out_size, void* d_ws, size_t ws_size,
                              hipStream_t stream) {
    (void)in_sizes; (void)n_in; (void)out_size; (void)ws_size;

    const float* x   = (const float*)d_in[0];
    const float* ew1 = (const float*)d_in[1];
    const float* eb1 = (const float*)d_in[2];
    const float* ew2 = (const float*)d_in[3];
    const float* eb2 = (const float*)d_in[4];
    const float* ew3 = (const float*)d_in[5];
    const float* eb3 = (const float*)d_in[6];
    const float* cb  = (const float*)d_in[7];
    const float* dw1 = (const float*)d_in[8];
    const float* db1 = (const float*)d_in[9];
    const float* dw2 = (const float*)d_in[10];
    const float* db2 = (const float*)d_in[11];
    const float* dw3 = (const float*)d_in[12];
    const float* db3 = (const float*)d_in[13];

    float* out   = (float*)d_out;
    float* lossp = out + (size_t)64 * 24576;   // x_recon is 1,572,864 floats
    float* zout  = lossp + 1;                  // then loss (1), then z (64*128)

    unsigned* zbits = (unsigned*)d_ws;                                  // 32 KB
    float*    g2buf = (float*)((char*)d_ws + 8192 * sizeof(unsigned));  // 16 KB

    vq_init<<<32, 256, 0, stream>>>(zbits, lossp);
    vq_encoder<<<dim3(64, 64), 256, 0, stream>>>(x, ew1, eb1, ew2, eb2, ew3, eb3, zbits);
    vq_assign_dec12<<<64, 256, 0, stream>>>(zbits, cb, dw1, db1, dw2, db2, zout, lossp, g2buf);
    vq_decoder3<<<768, 256, 0, stream>>>(g2buf, dw3, db3, out);
}
